// Encoder_Base_28509992911144
// MI455X (gfx1250) — matrix operand order repaired and checked by reading, not compiled, not run on hardware
//
#include <hip/hip_runtime.h>
#include <math.h>

typedef __attribute__((ext_vector_type(16))) _Float16 v16h;
typedef __attribute__((ext_vector_type(8)))  _Float16 v8h;
typedef __attribute__((ext_vector_type(8)))  float    v8f;

#define HDIM 128   // H
#define G4   512   // 4*H

// ---------------------------------------------------------------------------
// Kernel 0: convert W,U (f32, [k][n] with n in 0..511) to f16, N-major layout
// wt[n][k], ut[n][k]  (n = 0..511, k = 0..127). 256 KB total -> lives in L2.
// ---------------------------------------------------------------------------
__global__ void prep_weights_f16(const float* __restrict__ W,
                                 const float* __restrict__ U,
                                 _Float16* __restrict__ wt,
                                 _Float16* __restrict__ ut) {
    int idx = blockIdx.x * blockDim.x + threadIdx.x;
    if (idx < G4 * HDIM) {
        int n = idx >> 7;          // 0..511
        int k = idx & (HDIM - 1);  // 0..127
        wt[idx] = (_Float16)W[(size_t)k * G4 + n];
        ut[idx] = (_Float16)U[(size_t)k * G4 + n];
    }
}

__device__ __forceinline__ float hsig(float v) {
    return fminf(fmaxf(0.2f * v + 0.5f, 0.0f), 1.0f);
}

// Build a 16x32 f16 A-fragment from a global f32 row.
// ISA 16-bit A layout: lane holds row m = lane%16; halves [0..7] = K base..base+7,
// halves [8..15] = K base+16..base+23, where base = kk*32 + 8*(lane/16).
__device__ __forceinline__ v16h cvtA(const float* __restrict__ p, int b1) {
    float4 f0 = *(const float4*)(p + b1);
    float4 f1 = *(const float4*)(p + b1 + 4);
    float4 f2 = *(const float4*)(p + b1 + 16);
    float4 f3 = *(const float4*)(p + b1 + 20);
    v16h a;
    a[0]=(_Float16)f0.x;  a[1]=(_Float16)f0.y;  a[2]=(_Float16)f0.z;  a[3]=(_Float16)f0.w;
    a[4]=(_Float16)f1.x;  a[5]=(_Float16)f1.y;  a[6]=(_Float16)f1.z;  a[7]=(_Float16)f1.w;
    a[8]=(_Float16)f2.x;  a[9]=(_Float16)f2.y;  a[10]=(_Float16)f2.z; a[11]=(_Float16)f2.w;
    a[12]=(_Float16)f3.x; a[13]=(_Float16)f3.y; a[14]=(_Float16)f3.z; a[15]=(_Float16)f3.w;
    return a;
}

// ---------------------------------------------------------------------------
// Main fused kernel. 256 threads = 8 wave32s; wave w owns rows
// [blockIdx.x*128 + 16w, +16).  All matrix math on v_wmma_f32_16x16x32_f16.
// ---------------------------------------------------------------------------
__global__ __launch_bounds__(256)
void gru_fused_wmma(const float* __restrict__ x,
                    const float* __restrict__ h,
                    const _Float16* __restrict__ wt,   // [512][128] f16
                    const _Float16* __restrict__ ut,   // [512][128] f16
                    const float* __restrict__ bias,    // [512]
                    const int* __restrict__ x_mask,
                    const int* __restrict__ prev_has_value,
                    const int* __restrict__ has_value_tm1,
                    float* __restrict__ out_h,
                    float* __restrict__ out_hv) {
    __shared__ __align__(32) _Float16 lds_r1x[8][16][HDIM]; // (r1 * x) per-wave, row-major
    __shared__ __align__(32) _Float16 lds_r2h[8][16][HDIM]; // (r2 * h) per-wave
    __shared__ __align__(32) _Float16 lds_z  [8][16][HDIM]; // z gate  per-wave

    const int wv   = threadIdx.x >> 5;
    const int lane = threadIdx.x & 31;
    const int lg   = lane >> 4;   // lane half-group (0/1)
    const int ln   = lane & 15;   // n (B/C/D) or m (A) index
    const long rowbase = (long)blockIdx.x * 128 + wv * 16;

    // B-fragment: lane = column ncol, halves [0..7] = K base..base+7, [8..15] = K base+16..base+23, base = kk*32 + 8*lg.
    auto loadB = [&](const _Float16* __restrict__ w, int ncol, int kk) -> v16h {
        const _Float16* pb = w + (size_t)ncol * HDIM + kk * 32 + lg * 8;
        v8h b0 = *(const v8h*)pb;
        v8h b1 = *(const v8h*)(pb + 16);
        v16h r;
#pragma unroll
        for (int j = 0; j < 8; ++j) { r[j] = b0[j]; r[8 + j] = b1[j]; }
        return r;
    };

    // ---- A fragments (f32 -> f16) for x and h_tm1, K = 0..127 in 4 chunks ----
    v16h ax[4], ah[4];
    {
        const float* px = x + (rowbase + ln) * HDIM;
        const float* ph = h + (rowbase + ln) * HDIM;
#pragma unroll
        for (int kk = 0; kk < 4; ++kk) {
            int b1 = kk * 32 + lg * 8;
            ax[kk] = cvtA(px, b1);
            ah[kk] = cvtA(ph, b1);
        }
    }

    // ---- Phase 1: s = hard_sigmoid(x@W3 + h@U3 + b3), 24 N-tiles of 16 ----
#pragma unroll 1
    for (int nt = 0; nt < 24; ++nt) {
        const int col  = (nt & 7) * 16 + ln;  // column within a gate (0..127)
        const int gcol = nt * 16 + ln;        // column within [0,384)
        v8f acc = {};
#pragma unroll
        for (int kk = 0; kk < 4; ++kk) {
            v16h bw = loadB(wt, gcol, kk);
            acc = __builtin_amdgcn_wmma_f32_16x16x32_f16(false, ax[kk], false, bw,
                                                         (short)0, acc, false, false);
            v16h bu = loadB(ut, gcol, kk);
            acc = __builtin_amdgcn_wmma_f32_16x16x32_f16(false, ah[kk], false, bu,
                                                         (short)0, acc, false, false);
        }
        const float bv = bias[gcol];
        if (nt < 8) {            // z gate -> LDS (re-read in D-layout at the end)
#pragma unroll
            for (int i = 0; i < 8; ++i)
                lds_z[wv][i + 8 * lg][col] = (_Float16)hsig(acc[i] + bv);
        } else if (nt < 16) {    // r1 gate -> (r1 * x) staged row-major for phase-2 A
#pragma unroll
            for (int i = 0; i < 8; ++i) {
                float s  = hsig(acc[i] + bv);
                float xv = x[(rowbase + i + 8 * lg) * HDIM + col];
                lds_r1x[wv][i + 8 * lg][col] = (_Float16)(s * xv);
            }
        } else {                 // r2 gate -> (r2 * h)
#pragma unroll
            for (int i = 0; i < 8; ++i) {
                float s  = hsig(acc[i] + bv);
                float hv = h[(rowbase + i + 8 * lg) * HDIM + col];
                lds_r2h[wv][i + 8 * lg][col] = (_Float16)(s * hv);
            }
        }
    }
    // LDS data is wave-private (writer == reader wave): DScnt waits inserted by compiler.

    // ---- Phase 2 A fragments from LDS (row-major -> WMMA A layout) ----
    v16h ap[4], aq[4];
#pragma unroll
    for (int kk = 0; kk < 4; ++kk) {
        int b1 = kk * 32 + lg * 8;
        v8h p0 = *(const v8h*)&lds_r1x[wv][ln][b1];
        v8h p1 = *(const v8h*)&lds_r1x[wv][ln][b1 + 16];
        v8h q0 = *(const v8h*)&lds_r2h[wv][ln][b1];
        v8h q1 = *(const v8h*)&lds_r2h[wv][ln][b1 + 16];
        v16h a, c;
#pragma unroll
        for (int j = 0; j < 8; ++j) {
            a[j] = p0[j]; a[8 + j] = p1[j];
            c[j] = q0[j]; c[8 + j] = q1[j];
        }
        ap[kk] = a;
        aq[kk] = c;
    }

    // Row masks (uniform per 16-lane group; cached in L0)
    int hvb[8], mg[8];
#pragma unroll
    for (int i = 0; i < 8; ++i) {
        long r = rowbase + i + 8 * lg;
        hvb[i] = has_value_tm1[r];
        mg[i]  = x_mask[r] * prev_has_value[r];
    }

    // ---- Phase 2: h_cand = tanh((r1x)@W4 + (r2h)@U4 + b4); gate + merge ----
#pragma unroll 1
    for (int nt = 0; nt < 8; ++nt) {
        const int col = nt * 16 + ln;
        v8f acc = {};
#pragma unroll
        for (int kk = 0; kk < 4; ++kk) {
            v16h bw = loadB(wt, 384 + col, kk);
            acc = __builtin_amdgcn_wmma_f32_16x16x32_f16(false, ap[kk], false, bw,
                                                         (short)0, acc, false, false);
            v16h bu = loadB(ut, 384 + col, kk);
            acc = __builtin_amdgcn_wmma_f32_16x16x32_f16(false, aq[kk], false, bu,
                                                         (short)0, acc, false, false);
        }
        const float bv = bias[384 + col];
#pragma unroll
        for (int i = 0; i < 8; ++i) {
            long r = rowbase + i + 8 * lg;
            float hc    = tanhf(acc[i] + bv);
            float zv    = (float)lds_z[wv][i + 8 * lg][col];
            float hprev = h[r * HDIM + col];
            float xv    = x[r * HDIM + col];
            float h_    = zv * hprev + (1.0f - zv) * hc;
            float o = (hvb[i] != 0) ? h_ : xv;
            o       = (mg[i]  != 0) ? o  : hprev;
            out_h[r * HDIM + col] = o;
        }
    }

    // has_value: one row per (i, lg); lanes ln==0 of each half-group write 8 rows each.
    if (ln == 0) {
#pragma unroll
        for (int i = 0; i < 8; ++i) {
            long r = rowbase + i + 8 * lg;
            out_hv[r] = ((mg[i] != 0) || (hvb[i] != 0)) ? 1.0f : 0.0f;
        }
    }
}

// ---------------------------------------------------------------------------
extern "C" void kernel_launch(void* const* d_in, const int* in_sizes, int n_in,
                              void* d_out, int out_size, void* d_ws, size_t ws_size,
                              hipStream_t stream) {
    const float* x       = (const float*)d_in[0];
    const float* h_tm1   = (const float*)d_in[1];
    const float* W       = (const float*)d_in[2];
    const float* U       = (const float*)d_in[3];
    const float* b       = (const float*)d_in[4];
    const int*   x_mask  = (const int*)d_in[5];
    const int*   prev_hv = (const int*)d_in[6];
    const int*   hv_tm1  = (const int*)d_in[7];

    const long B = (long)in_sizes[0] / HDIM;

    _Float16* wt = (_Float16*)d_ws;              // 512*128 f16 = 128 KB
    _Float16* ut = wt + (size_t)G4 * HDIM;       // next 128 KB

    float* out_h  = (float*)d_out;               // B*128 floats
    float* out_hv = out_h + (size_t)B * HDIM;    // B floats (bool as 0.0/1.0)

    hipLaunchKernelGGL(prep_weights_f16, dim3((G4 * HDIM + 255) / 256), dim3(256), 0, stream,
                       W, U, wt, ut);
    hipLaunchKernelGGL(gru_fused_wmma, dim3((unsigned)(B / 128)), dim3(256), 0, stream,
                       x, h_tm1, wt, ut, b, x_mask, prev_hv, hv_tm1, out_h, out_hv);
}
